// WindowAttention_57208964383233
// MI455X (gfx1250) — matrix operand order repaired and checked by reading, not compiled, not run on hardware
//
#include <hip/hip_runtime.h>

typedef _Float16 v16h __attribute__((ext_vector_type(16)));
typedef _Float16 v8h  __attribute__((ext_vector_type(8)));
typedef float    v8f  __attribute__((ext_vector_type(8)));
typedef float    v4f  __attribute__((ext_vector_type(4)));

#define WMMA_F16(A, B, C) \
    __builtin_amdgcn_wmma_f32_16x16x32_f16(false, (A), false, (B), (short)0, (C), false, false)

// ---------------- fragment helpers (layouts per CDNA5 ISA 7.12.2) ----------------

// A-fragment (16x32 f16, MxK) loaded from f16 row-major memory.
// rowptr = &mat[row][0]; halves 0..7 <- K = k0 + (lane>=16?8:0) + e ; halves 8..15 at +16.
__device__ __forceinline__ v16h a_frag_f16(const _Float16* rowptr, int k0, int lane) {
    const int kb = k0 + ((lane & 16) ? 8 : 0);
    const _Float16* p = rowptr + kb;
    v16h a;
    *((v8h*)&a)       = *(const v8h*)(p);
    *(((v8h*)&a) + 1) = *(const v8h*)(p + 16);
    return a;
}

// A-fragment from f32 row-major global memory with f32->f16 convert; zero if !valid.
__device__ __forceinline__ v16h a_frag_f32(const float* base, int ld, int row, int k0,
                                           int lane, bool valid) {
    v16h a;
    if (valid) {
        const int kb = k0 + ((lane & 16) ? 8 : 0);
        const v4f* p0 = (const v4f*)(base + row * ld + kb);
        const v4f* p1 = (const v4f*)(base + row * ld + kb + 16);
        v4f x0 = p0[0], x1 = p0[1], x2 = p1[0], x3 = p1[1];
#pragma unroll
        for (int i = 0; i < 4; ++i) {
            a[i]      = (_Float16)x0[i];
            a[i + 4]  = (_Float16)x1[i];
            a[i + 8]  = (_Float16)x2[i];
            a[i + 12] = (_Float16)x3[i];
        }
    } else {
#pragma unroll
        for (int i = 0; i < 16; ++i) a[i] = (_Float16)0.0f;
    }
    return a;
}

// B-fragment (32x16 f16, KxN) from "BT" storage: row = N, K contiguous (ld in halves).
// lane n: N = n0 + n%16 ; halves 0..7 <- K = k0 + (n>=16?8:0) + e ; halves 8..15 at +16.
__device__ __forceinline__ v16h b_frag_f16(const _Float16* base, int ld, int n0, int k0,
                                           int lane) {
    const int n  = n0 + (lane & 15);
    const int kb = k0 + ((lane & 16) ? 8 : 0);
    const _Float16* p = base + n * ld + kb;
    v16h b;
    *((v8h*)&b)       = *(const v8h*)(p);
    *(((v8h*)&b) + 1) = *(const v8h*)(p + 16);
    return b;
}

// ---------------- prep: f16 weights + gathered bias table ----------------

__global__ void swin_prep_kernel(const float* __restrict__ qkv_w,
                                 const float* __restrict__ proj_w,
                                 const float* __restrict__ bias_table,
                                 const int*   __restrict__ rel_index,
                                 _Float16* __restrict__ qkvw16,
                                 _Float16* __restrict__ projw16,
                                 float* __restrict__ biasf) {
    int idx = blockIdx.x * blockDim.x + threadIdx.x;
    if (idx < 576 * 192) { qkvw16[idx] = (_Float16)qkv_w[idx]; return; }
    idx -= 576 * 192;
    if (idx < 192 * 192) { projw16[idx] = (_Float16)proj_w[idx]; return; }
    idx -= 192 * 192;
    if (idx < 6 * 49 * 49) {
        const int h  = idx / 2401;
        const int ij = idx % 2401;
        biasf[idx] = bias_table[rel_index[ij] * 6 + h];  // [6][49][49]
    }
}

// ---------------- fused window attention: 1 block = 1 window, wave = head ----------------

__global__ __launch_bounds__(192) void swin_win_attn(
    const float* __restrict__ x,        // [4096][49][192]
    const float* __restrict__ qkv_b,    // [576]
    const float* __restrict__ proj_b,   // [192]
    const _Float16* __restrict__ qkvw,  // [576][192] f16  (B^T layout for QKV GEMM)
    const _Float16* __restrict__ projw, // [192][192] f16  (B^T layout for proj GEMM)
    const float* __restrict__ biasf,    // [6][49][49]
    float* __restrict__ out)            // [4096][49][192]
{
    extern __shared__ _Float16 smem[];
    const int lane = threadIdx.x & 31;
    const int h    = threadIdx.x >> 5;        // wave id == head id (6 waves)
    const int l16  = lane & 15;
    const int hi8  = (lane & 16) ? 8 : 0;
    const int win  = blockIdx.x;
    const float* xw = x + (size_t)win * (49 * 192);

    // Per-head LDS: Q[64][32], K[64][32], VT[32][64] (f16); P[64][64] overlays Q+K.
    _Float16* sQ  = smem + h * 6144;
    _Float16* sK  = sQ + 2048;
    _Float16* sVT = sQ + 4096;
    _Float16* sP  = sQ;
    _Float16* sO  = smem + 6 * 6144;          // shared O [64][192] f16

    const float scale = 0.17677669529663687f; // 1/sqrt(32), folded into q

    // -------- Phase 1: QKV projection for this head --------
    for (int mi = 0; mi < 4; ++mi) {
        const int arow = mi * 16 + l16;
        const bool avalid = arow < 49;
        v16h a[6];
#pragma unroll
        for (int kt = 0; kt < 6; ++kt)
            a[kt] = a_frag_f32(xw, 192, arow, kt * 32, lane, avalid);
#pragma unroll
        for (int s = 0; s < 3; ++s) {
#pragma unroll
            for (int nj = 0; nj < 2; ++nj) {
                const int c0 = s * 192 + h * 32 + nj * 16;
                v8f acc = {};
#pragma unroll
                for (int kt = 0; kt < 6; ++kt) {
                    v16h b = b_frag_f16(qkvw, 192, c0, kt * 32, lane);
                    acc = WMMA_F16(a[kt], b, acc);
                }
                const float bc = qkv_b[c0 + l16];
                if (s == 0) {            // Q: token-major [64][32], pre-scaled
#pragma unroll
                    for (int r = 0; r < 8; ++r)
                        sQ[(mi * 16 + hi8 + r) * 32 + nj * 16 + l16] =
                            (_Float16)((acc[r] + bc) * scale);
                } else if (s == 1) {     // K: token-major [64][32]
#pragma unroll
                    for (int r = 0; r < 8; ++r)
                        sK[(mi * 16 + hi8 + r) * 32 + nj * 16 + l16] =
                            (_Float16)(acc[r] + bc);
                } else {                 // V: transposed [32][64] (contiguous token stores)
                    v8h t;
#pragma unroll
                    for (int r = 0; r < 8; ++r) t[r] = (_Float16)(acc[r] + bc);
                    *(v8h*)(sVT + (nj * 16 + l16) * 64 + mi * 16 + hi8) = t;
                }
            }
        }
    }
    // Same-wave LDS RAW: DS ops are in-order per wave -> no barrier needed.

    // -------- Phase 2: attention for this head --------
    v16h qa[4], kf[4];
#pragma unroll
    for (int mi = 0; mi < 4; ++mi)
        qa[mi] = a_frag_f16(sQ + (mi * 16 + l16) * 32, 0, lane);
#pragma unroll
    for (int nj = 0; nj < 4; ++nj)
        kf[nj] = b_frag_f16(sK, 32, nj * 16, 0, lane);

    v8f c[4][4];
#pragma unroll
    for (int mi = 0; mi < 4; ++mi)
#pragma unroll
        for (int nj = 0; nj < 4; ++nj) {
            v8f z = {};
            c[mi][nj] = WMMA_F16(qa[mi], kf[nj], z);   // logits (q pre-scaled)
        }

    // bias + mask + softmax; D layout: a row's 16 cols live in one half-wave.
    const float* bh = biasf + h * 2401;
#pragma unroll
    for (int mi = 0; mi < 4; ++mi) {
#pragma unroll
        for (int r = 0; r < 8; ++r) {
            const int m = mi * 16 + hi8 + r;
            float v0[4];
#pragma unroll
            for (int nj = 0; nj < 4; ++nj) {
                const int col = nj * 16 + l16;
                float t = c[mi][nj][r];
                if (col < 49) {
                    if (m < 49) t += bh[m * 49 + col];
                } else {
                    t = -1e30f;          // mask padded key columns
                }
                v0[nj] = t;
            }
            float mx = fmaxf(fmaxf(v0[0], v0[1]), fmaxf(v0[2], v0[3]));
#pragma unroll
            for (int d = 1; d < 16; d <<= 1) mx = fmaxf(mx, __shfl_xor(mx, d, 32));
            float sum = 0.0f;
#pragma unroll
            for (int nj = 0; nj < 4; ++nj) { v0[nj] = __expf(v0[nj] - mx); sum += v0[nj]; }
#pragma unroll
            for (int d = 1; d < 16; d <<= 1) sum += __shfl_xor(sum, d, 32);
            const float inv = 1.0f / sum;
#pragma unroll
            for (int nj = 0; nj < 4; ++nj)
                sP[m * 64 + nj * 16 + l16] = (_Float16)(v0[nj] * inv);
        }
    }

    // P @ V  -> shared O
#pragma unroll
    for (int njd = 0; njd < 2; ++njd) {
        v16h vb0 = b_frag_f16(sVT, 64, njd * 16, 0, lane);
        v16h vb1 = b_frag_f16(sVT, 64, njd * 16, 32, lane);
#pragma unroll
        for (int mi = 0; mi < 4; ++mi) {
            v16h p0 = a_frag_f16(sP + (mi * 16 + l16) * 64, 0, lane);
            v16h p1 = a_frag_f16(sP + (mi * 16 + l16) * 64, 32, lane);
            v8f o = {};
            o = WMMA_F16(p0, vb0, o);
            o = WMMA_F16(p1, vb1, o);
            const int dcol = h * 32 + njd * 16 + l16;
#pragma unroll
            for (int r = 0; r < 8; ++r)
                sO[(mi * 16 + hi8 + r) * 192 + dcol] = (_Float16)o[r];
        }
    }

    __syncthreads();   // all heads' O slices must be visible to all waves

    // -------- Phase 3: output projection (2 col-tiles per wave) --------
    float* outw = out + (size_t)win * (49 * 192);
    for (int mi = 0; mi < 4; ++mi) {
        v16h oa[6];
#pragma unroll
        for (int kt = 0; kt < 6; ++kt)
            oa[kt] = a_frag_f16(sO + (mi * 16 + l16) * 192, kt * 32, lane);
#pragma unroll
        for (int njl = 0; njl < 2; ++njl) {
            const int c0 = (h * 2 + njl) * 16;
            v8f acc = {};
#pragma unroll
            for (int kt = 0; kt < 6; ++kt) {
                v16h b = b_frag_f16(projw, 192, c0, kt * 32, lane);
                acc = WMMA_F16(oa[kt], b, acc);
            }
            const float pb = proj_b[c0 + l16];
#pragma unroll
            for (int r = 0; r < 8; ++r) {
                const int m = mi * 16 + hi8 + r;
                if (m < 49) outw[m * 192 + c0 + l16] = acc[r] + pb;
            }
        }
    }
}

// ---------------- host launch ----------------

extern "C" void kernel_launch(void* const* d_in, const int* in_sizes, int n_in,
                              void* d_out, int out_size, void* d_ws, size_t ws_size,
                              hipStream_t stream) {
    const float* x          = (const float*)d_in[0];
    const float* qkv_w      = (const float*)d_in[1];
    const float* qkv_b      = (const float*)d_in[2];
    const float* proj_w     = (const float*)d_in[3];
    const float* proj_b     = (const float*)d_in[4];
    const float* bias_table = (const float*)d_in[5];
    const int*   rel_index  = (const int*)d_in[6];
    float* out = (float*)d_out;

    // d_ws layout: qkv_w f16 (221184 B) | proj_w f16 (73728 B) | bias f32 (57624 B)
    _Float16* qkvw16  = (_Float16*)d_ws;
    _Float16* projw16 = (_Float16*)((char*)d_ws + 221184);
    float*    biasf   = (float*)((char*)d_ws + 294912);

    const int prep_items = 576 * 192 + 192 * 192 + 6 * 49 * 49;
    swin_prep_kernel<<<(prep_items + 255) / 256, 256, 0, stream>>>(
        qkv_w, proj_w, bias_table, rel_index, qkvw16, projw16, biasf);

    const int smem_bytes = 98304;  // 6 heads * 12KB + 24KB shared O
    hipFuncSetAttribute((const void*)swin_win_attn,
                        hipFuncAttributeMaxDynamicSharedMemorySize, smem_bytes);
    swin_win_attn<<<4096, 192, smem_bytes, stream>>>(
        x, qkv_b, proj_b, qkvw16, projw16, biasf, out);
}
